// MetaCognitionModule_21096879358445
// MI455X (gfx1250) — matrix operand order repaired and checked by reading, compiled, not run on hardware
//
#include <hip/hip_runtime.h>
#include <hip/hip_bf16.h>
#include <math.h>

// MetaCognitionModule for MI455X (gfx1250).
// Compute-bound (~0.46 TFLOP vs ~0.6 GB traffic) -> bf16 WMMA 16x16x32, f32 acc.
// Evaluator GEMM split: concat([x,hyps])@We1 == x@We1[:H] + hyps@We1[H:].
// GEMM is software-pipelined with double-buffered LDS: A tiles staged via
// gfx1250 async global->LDS copies (ASYNCcnt), B tiles via register transpose
// whose LDS stores are scheduled after the WMMA issue.

static constexpr int Hn = 1024;
static constexpr int Kn = 8;
static constexpr int Bn = 8192;

typedef __attribute__((ext_vector_type(16))) __bf16 v16bf;
typedef __attribute__((ext_vector_type(8)))  float  v8f;

// ---------------- f32 -> bf16 convert ----------------
__global__ void k_f2bf(const float* __restrict__ in, __bf16* __restrict__ out, long n) {
  long i = (long)blockIdx.x * blockDim.x + threadIdx.x;
  long stride = (long)gridDim.x * blockDim.x;
  for (; i < n; i += stride) out[i] = (__bf16)in[i];
}

// ---------------- gfx1250 async global->LDS copy ----------------
__device__ inline void async_b128_to_lds(uint32_t lds_addr, const void* gptr) {
  asm volatile("global_load_async_to_lds_b128 %0, %1, off"
               :: "v"(lds_addr), "v"((unsigned long long)(uintptr_t)gptr)
               : "memory");
}
__device__ inline void wait_asynccnt0() {
  asm volatile("s_wait_asynccnt 0" ::: "memory");
}

// ---------------- WMMA fragment loaders (contiguous k order, same for A and B) ----------------
// A 16x32 bf16: lane L holds row M=L%16; elem i -> K=(L/16)*16+i.
// LDS A is [128][32] row-major.
__device__ inline v16bf frag_a(const __bf16* sA, int m_tile, int lane) {
  int row = m_tile * 16 + (lane & 15);
  int kb  = (lane >> 4) * 16;
  union { v16bf v; uint4 u[2]; } f;
  f.u[0] = *(const uint4*)(sA + row * 32 + kb);
  f.u[1] = *(const uint4*)(sA + row * 32 + kb + 8);
  return f.v;
}
// B 32x16 bf16: lane L holds col N=L%16; elem i -> K=(L/16)*16+i.
// LDS B is stored transposed [128][32] (N-major) so elems are contiguous.
__device__ inline v16bf frag_b(const __bf16* sBT, int n_tile, int lane) {
  int col = n_tile * 16 + (lane & 15);
  int kb  = (lane >> 4) * 16;
  union { v16bf v; uint4 u[2]; } f;
  const uint4* p = (const uint4*)(sBT + col * 32 + kb);
  f.u[0] = p[0];
  f.u[1] = p[1];
  return f.v;
}

// B tile register transpose helpers
__device__ inline void store_b_tile_lds(__bf16* sBT, int brow, int bcol,
                                        uint4 u0, uint4 u1) {
  union { uint4 u[2]; __bf16 h[16]; } tb;
  tb.u[0] = u0; tb.u[1] = u1;
#pragma unroll
  for (int j = 0; j < 16; j++) sBT[(bcol + j) * 32 + brow] = tb.h[j];
}

// ---------------- tiled bf16 WMMA GEMM (double-buffered pipeline) ----------------
// C[M,N] = A[M,Kd] @ Bm[Kd,N] (+bias[N]) (+addrow[row>>rowshift, N]); f32 out,
// optional bf16 mirror. Block tile 128x128, 256 threads = 8 waves (4M x 2N),
// each wave computes 32x64 via 2x4 WMMA tiles.
__global__ __launch_bounds__(256) void k_gemm_bf16(
    const __bf16* __restrict__ A, const __bf16* __restrict__ Bm,
    const float* __restrict__ bias, const float* __restrict__ addrow, int rowshift,
    float* __restrict__ C, __bf16* __restrict__ Cbf,
    int M, int N, int Kd) {
  __shared__ __align__(16) __bf16 sA[2][128 * 32];
  __shared__ __align__(16) __bf16 sBT[2][128 * 32];

  const int t    = threadIdx.x;
  const int lane = t & 31;
  const int w    = t >> 5;
  const int bm   = blockIdx.y;
  const int bn   = blockIdx.x;
  const int wm   = w >> 1;   // 0..3 -> 32-row band
  const int wn   = w & 1;    // 0..1 -> 64-col band

  v8f acc[2][4];
  v8f zero = {0.f, 0.f, 0.f, 0.f, 0.f, 0.f, 0.f, 0.f};
#pragma unroll
  for (int i = 0; i < 2; i++)
#pragma unroll
    for (int j = 0; j < 4; j++) acc[i][j] = zero;

  const int arow = t >> 1, ahalf = (t & 1) * 16;   // A: 128 rows x 2 halves
  const int brow = t >> 3, bcol = (t & 7) * 16;    // B: 32 rows x 8 col-chunks

  const __bf16* agp = A + (size_t)(bm * 128 + arow) * Kd + ahalf;      // +k0
  const __bf16* bgp = Bm + (size_t)brow * N + bn * 128 + bcol;         // +k0*N
  const size_t  bstep = (size_t)32 * N;

  uint32_t aslds[2][2];
#pragma unroll
  for (int s = 0; s < 2; s++) {
    __bf16* as = sA[s] + arow * 32 + ahalf;
    aslds[s][0] = (uint32_t)(uintptr_t)as;
    aslds[s][1] = (uint32_t)(uintptr_t)(as + 8);
  }

  // prologue: stage tile 0 into buffer 0
  async_b128_to_lds(aslds[0][0], agp);
  async_b128_to_lds(aslds[0][1], agp + 8);
  {
    const uint4* g = (const uint4*)bgp;
    store_b_tile_lds(sBT[0], brow, bcol, g[0], g[1]);
  }
  wait_asynccnt0();
  __syncthreads();

  int buf = 0;
  for (int k0 = 0; k0 < Kd; k0 += 32) {
    const int  nbuf = buf ^ 1;
    const bool has_next = (k0 + 32) < Kd;

    // start tile k+1: async A copies + B global loads into registers
    uint4 bu0, bu1;
    if (has_next) {
      async_b128_to_lds(aslds[nbuf][0], agp + k0 + 32);
      async_b128_to_lds(aslds[nbuf][1], agp + k0 + 40);
      const uint4* g = (const uint4*)(bgp + bstep + (size_t)k0 * N);
      bu0 = g[0]; bu1 = g[1];
      __builtin_prefetch(agp + k0 + 64, 0, 3);                 // warm L2 for k+2
      __builtin_prefetch(bgp + 2 * bstep + (size_t)k0 * N, 0, 3);
    }

    // compute tile k from current buffer
    v16bf af[2], bfr[4];
#pragma unroll
    for (int i = 0; i < 2; i++) af[i] = frag_a(sA[buf], wm * 2 + i, lane);
#pragma unroll
    for (int j = 0; j < 4; j++) bfr[j] = frag_b(sBT[buf], wn * 4 + j, lane);
#pragma unroll
    for (int i = 0; i < 2; i++)
#pragma unroll
      for (int j = 0; j < 4; j++)
        acc[i][j] = __builtin_amdgcn_wmma_f32_16x16x32_bf16(
            false, af[i], false, bfr[j], (short)0, acc[i][j], false, false);

    // finish tile k+1: transpose-store B into next buffer (behind WMMA issue)
    if (has_next) store_b_tile_lds(sBT[nbuf], brow, bcol, bu0, bu1);

    wait_asynccnt0();   // next-tile async A done (dscnt wait added by compiler)
    __syncthreads();
    buf = nbuf;
  }

  // Epilogue. D layout: VGPR r, lane L -> row r + (L/16)*8, col L%16.
#pragma unroll
  for (int i = 0; i < 2; i++) {
#pragma unroll
    for (int j = 0; j < 4; j++) {
      const int gcol = bn * 128 + wn * 64 + j * 16 + (lane & 15);
      const float bv = bias ? bias[gcol] : 0.f;
#pragma unroll
      for (int r = 0; r < 8; r++) {
        int grow = bm * 128 + wm * 32 + i * 16 + r + ((lane >> 4) * 8);
        float v = acc[i][j][r] + bv;
        if (addrow) v += addrow[(size_t)(grow >> rowshift) * N + gcol];
        size_t idx = (size_t)grow * N + gcol;
        C[idx] = v;
        if (Cbf) Cbf[idx] = (__bf16)v;
      }
    }
  }
}

// ---------------- LayerNorm + ReLU -> bf16 (block per row) ----------------
__global__ __launch_bounds__(256) void k_ln_relu_bf16(
    const float* __restrict__ t, const float* __restrict__ g,
    const float* __restrict__ beta, __bf16* __restrict__ out, int cols) {
  __shared__ float r1[256], r2[256];
  const int row = blockIdx.x;
  const float* rp = t + (size_t)row * cols;
  float s = 0.f, s2 = 0.f;
  for (int c = threadIdx.x; c < cols; c += 256) { float v = rp[c]; s += v; s2 += v * v; }
  r1[threadIdx.x] = s; r2[threadIdx.x] = s2;
  __syncthreads();
  for (int off = 128; off > 0; off >>= 1) {
    if (threadIdx.x < off) { r1[threadIdx.x] += r1[threadIdx.x + off];
                             r2[threadIdx.x] += r2[threadIdx.x + off]; }
    __syncthreads();
  }
  float mean = r1[0] / cols;
  float inv  = rsqrtf(r2[0] / cols - mean * mean + 1e-5f);
  for (int c = threadIdx.x; c < cols; c += 256) {
    float v = (rp[c] - mean) * inv * g[c] + beta[c];
    out[(size_t)row * cols + c] = (__bf16)fmaxf(v, 0.f);
  }
}

// ---------------- LN + ReLU + dot(We2) + sigmoid (block per row) ----------------
__global__ __launch_bounds__(256) void k_eval_score(
    const float* __restrict__ ev, const float* __restrict__ ge,
    const float* __restrict__ betae, const float* __restrict__ We2,
    const float* __restrict__ be2, float* __restrict__ scores, int cols) {
  __shared__ float r1[256], r2[256];
  const int row = blockIdx.x;
  const float* rp = ev + (size_t)row * cols;
  float s = 0.f, s2 = 0.f;
  for (int c = threadIdx.x; c < cols; c += 256) { float v = rp[c]; s += v; s2 += v * v; }
  r1[threadIdx.x] = s; r2[threadIdx.x] = s2;
  __syncthreads();
  for (int off = 128; off > 0; off >>= 1) {
    if (threadIdx.x < off) { r1[threadIdx.x] += r1[threadIdx.x + off];
                             r2[threadIdx.x] += r2[threadIdx.x + off]; }
    __syncthreads();
  }
  float mean = r1[0] / cols;
  float inv  = rsqrtf(r2[0] / cols - mean * mean + 1e-5f);
  __syncthreads();
  float dot = 0.f;
  for (int c = threadIdx.x; c < cols; c += 256) {
    float e = fmaxf((rp[c] - mean) * inv * ge[c] + betae[c], 0.f);
    dot += e * We2[c];
  }
  r1[threadIdx.x] = dot;
  __syncthreads();
  for (int off = 128; off > 0; off >>= 1) {
    if (threadIdx.x < off) r1[threadIdx.x] += r1[threadIdx.x + off];
    __syncthreads();
  }
  if (threadIdx.x == 0) {
    float z = r1[0] + be2[0];
    scores[row] = 1.f / (1.f + __expf(-z));
  }
}

// ---------------- argmax over K + gather best hypothesis ----------------
__global__ __launch_bounds__(256) void k_select(
    const float* __restrict__ scores, const float* __restrict__ hyps,
    float* __restrict__ best_hyp, float* __restrict__ best_idx,
    float* __restrict__ best_sc) {
  const int b = blockIdx.x;
  __shared__ int sidx;
  if (threadIdx.x == 0) {
    int bi = 0; float bs = scores[(size_t)b * Kn];
    for (int k = 1; k < Kn; k++) {
      float v = scores[(size_t)b * Kn + k];
      if (v > bs) { bs = v; bi = k; }   // strict > keeps first max (argmax semantics)
    }
    sidx = bi;
    ((int*)best_idx)[b] = bi;
    best_sc[b] = bs;
  }
  __syncthreads();
  const int bi = sidx;
  const float* src = hyps + ((size_t)b * Kn + bi) * Hn;
  float* dst = best_hyp + (size_t)b * Hn;
  for (int c = threadIdx.x; c < Hn; c += 256) dst[c] = src[c];
}

extern "C" void kernel_launch(void* const* d_in, const int* in_sizes, int n_in,
                              void* d_out, int out_size, void* d_ws, size_t ws_size,
                              hipStream_t stream) {
  (void)in_sizes; (void)n_in; (void)out_size; (void)ws_size;
  const float* x     = (const float*)d_in[0];
  const float* W1    = (const float*)d_in[1];
  const float* b1    = (const float*)d_in[2];
  const float* g1    = (const float*)d_in[3];
  const float* beta1 = (const float*)d_in[4];
  const float* W2    = (const float*)d_in[5];
  const float* b2    = (const float*)d_in[6];
  const float* We1   = (const float*)d_in[7];
  const float* be1   = (const float*)d_in[8];
  const float* ge    = (const float*)d_in[9];
  const float* betae = (const float*)d_in[10];
  const float* We2   = (const float*)d_in[11];
  const float* be2   = (const float*)d_in[12];

  // Output tuple, flat: best_hyp [B,H] | hyps [B,K,H] | scores [B,K] | best_idx [B,1] | best_scores [B,1]
  float* out      = (float*)d_out;
  float* best_hyp = out;
  float* hyps     = best_hyp + (size_t)Bn * Hn;
  float* scores   = hyps + (size_t)Bn * Kn * Hn;
  float* best_idx = scores + (size_t)Bn * Kn;
  float* best_sc  = best_idx + Bn;

  // Workspace layout (256B aligned).
  char* ws = (char*)d_ws;
  size_t o = 0;
  auto salloc = [&](size_t bytes) -> void* {
    void* p = ws + o;
    o += (bytes + 255) & ~(size_t)255;
    return p;
  };
  __bf16* xbf   = (__bf16*)salloc((size_t)Bn * Hn * 2);
  __bf16* w1bf  = (__bf16*)salloc((size_t)Hn * 2 * Hn * 2);
  __bf16* w2bf  = (__bf16*)salloc((size_t)2 * Hn * Kn * Hn * 2);
  __bf16* we1bf = (__bf16*)salloc((size_t)2 * Hn * Hn * 2);
  float*  tbuf  = (float*) salloc((size_t)Bn * 2 * Hn * 4);
  __bf16* hbf   = (__bf16*)salloc((size_t)Bn * 2 * Hn * 2);
  __bf16* hypbf = (__bf16*)salloc((size_t)Bn * Kn * Hn * 2);
  float*  xe    = (float*) salloc((size_t)Bn * Hn * 4);
  float*  ev    = (float*) salloc((size_t)Bn * Kn * Hn * 4);

  // 1) precision-convert operands once
  k_f2bf<<<2048, 256, 0, stream>>>(x,   xbf,   (long)Bn * Hn);
  k_f2bf<<<2048, 256, 0, stream>>>(W1,  w1bf,  (long)Hn * 2 * Hn);
  k_f2bf<<<4096, 256, 0, stream>>>(W2,  w2bf,  (long)2 * Hn * Kn * Hn);
  k_f2bf<<<2048, 256, 0, stream>>>(We1, we1bf, (long)2 * Hn * Hn);

  // 2) gen1: t = x@W1 + b1   [B, 2H]
  k_gemm_bf16<<<dim3(2 * Hn / 128, Bn / 128), 256, 0, stream>>>(
      xbf, w1bf, b1, nullptr, 0, tbuf, nullptr, Bn, 2 * Hn, Hn);
  // 3) h = bf16(relu(LN(t)))
  k_ln_relu_bf16<<<Bn, 256, 0, stream>>>(tbuf, g1, beta1, hbf, 2 * Hn);
  // 4) gen2: hyps = h@W2 + b2 (f32 to output, bf16 mirror for evaluator)
  k_gemm_bf16<<<dim3(Kn * Hn / 128, Bn / 128), 256, 0, stream>>>(
      hbf, w2bf, b2, nullptr, 0, hyps, hypbf, Bn, Kn * Hn, 2 * Hn);
  // 5) xe = x@We1[:H] + be1   [B, H]  (shared across all K hypotheses)
  k_gemm_bf16<<<dim3(Hn / 128, Bn / 128), 256, 0, stream>>>(
      xbf, we1bf, be1, nullptr, 0, xe, nullptr, Bn, Hn, Hn);
  // 6) ev = hyps@We1[H:] + xe[row>>3]   [B*K, H]
  k_gemm_bf16<<<dim3(Hn / 128, Bn * Kn / 128), 256, 0, stream>>>(
      hypbf, we1bf + (size_t)Hn * Hn, nullptr, xe, 3, ev, nullptr, Bn * Kn, Hn, Hn);
  // 7) scores = sigmoid(relu(LN(ev)) @ We2 + be2)
  k_eval_score<<<Bn * Kn, 256, 0, stream>>>(ev, ge, betae, We2, be2, scores, Hn);
  // 8) argmax + gather
  k_select<<<Bn, 256, 0, stream>>>(scores, hyps, best_hyp, best_idx, best_sc);
}
